// Rainbow_82111184765533
// MI455X (gfx1250) — matrix operand order repaired and checked by reading, not compiled, not run on hardware
//
#include <hip/hip_runtime.h>

// ---------------------------------------------------------------------------
// Rainbow-DQN distributional loss on MI455X (gfx1250).
//  * One-time f32->bf16 conversion of inputs + weights.
//  * GEMMs: v_wmma_f32_16x16x32_bf16, double-buffered LDS tiles fed by
//    global_load_async_to_lds_b128 (ASYNCcnt-tracked, no VGPR round trip).
//  * Distributional tail split into 3 small per-row kernels so the val/adv
//    buffers are reused across the three forward passes.
// ---------------------------------------------------------------------------

typedef __bf16 bf16;
typedef __attribute__((ext_vector_type(16))) __bf16 v16bf;
typedef __attribute__((ext_vector_type(8)))  __bf16 v8bf;
typedef __attribute__((ext_vector_type(4)))  __bf16 v4bf;
typedef __attribute__((ext_vector_type(8)))  float  v8f;

#define ATOMS   51
#define NACT    16
#define OA      (NACT * ATOMS)   // 816
#define BSZ     4096
#define DIN     2048
#define HID     1024
#define VAL_LD  64               // padded ld for the 51-wide value head

// ------------------------- f32 -> bf16 conversion --------------------------
__global__ __launch_bounds__(256)
void cvt_f32_to_bf16(const float* __restrict__ in, bf16* __restrict__ out, int n4) {
  const int i = blockIdx.x * 256 + threadIdx.x;
  if (i < n4) {
    const float4 f = ((const float4*)in)[i];
    v4bf o;
    o[0] = (bf16)f.x; o[1] = (bf16)f.y; o[2] = (bf16)f.z; o[3] = (bf16)f.w;
    ((v4bf*)out)[i] = o;
  }
}

// ------------------------------ GEMM ---------------------------------------
// C[M,N] = act(A[M,K](bf16) * W[N,K](bf16)^T + bias[N](f32)); M multiple of BM.
constexpr int BM = 128, BN = 64, BK = 64;
constexpr int LDTA = BK + 8;                 // 72 bf16 = 144B row (16B aligned, bank-skewed)
constexpr int A_TILE_B = BM * LDTA * 2;      // 18432 B
constexpr int W_TILE_B = BN * LDTA * 2;      // 9216 B

union FragU { v16bf v; v8bf h[2]; };

// A fragment (16x32, 16-bit): lane L -> row (L&15); regs0..3 = K half*8..+7,
// regs4..7 = K 16+half*8..+7  (ISA 7.12.2 "16-bit A-Matrix 16x32").
__device__ inline v16bf load_fragA(const bf16* row, int half) {
  FragU u;
  const int kb = half * 8;
  u.h[0] = *(const v8bf*)(row + kb);
  u.h[1] = *(const v8bf*)(row + 16 + kb);
  return u.v;
}
// B fragment (32x16, 16-bit): lane L -> column (L&15); regs0..3 = K half*8..+7,
// regs4..7 = K 16+half*8..+7. Column n of B == row n of W (row-major [N,K]).
__device__ inline v16bf load_fragB(const bf16* row, int half) {
  FragU u;
  const int kb = half * 8;
  u.h[0] = *(const v8bf*)(row + kb);
  u.h[1] = *(const v8bf*)(row + 16 + kb);
  return u.v;
}

// async 16B global -> LDS copy (per-lane), tracked with ASYNCcnt
__device__ inline void async_cp16(unsigned lds_addr, unsigned long long gaddr) {
  asm volatile("global_load_async_to_lds_b128 %0, %1, off"
               :: "v"(lds_addr), "v"(gaddr) : "memory");
}

template <typename TOut, bool RELU>
__global__ __launch_bounds__(256)
void gemm_bf16_async(const bf16* __restrict__ A, int lda,
                     const bf16* __restrict__ W,      // [N,K] row-major
                     const float* __restrict__ bias,  // [N]
                     TOut* __restrict__ C, int ldc,
                     int N, int K) {
  __shared__ bf16 sA[2][BM * LDTA];
  __shared__ bf16 sW[2][BN * LDTA];

  const int tid   = threadIdx.x;
  const int wave  = tid >> 5, lane = tid & 31;
  const int wm    = wave & 3;           // 4 waves along M (4*32 = 128)
  const int wn    = wave >> 2;          // 2 waves along N (2*32 = 64)
  const int m0    = blockIdx.y * BM;
  const int n0    = blockIdx.x * BN;
  const int lrow  = lane & 15;
  const int lhalf = lane >> 4;

  // per-thread async-copy chunk coordinates (16B granules)
  // A tile: 128 rows x 8 chunks = 1024 -> 4 per thread
  // W tile:  64 rows x 8 chunks =  512 -> 2 per thread
  const unsigned sA0 = (unsigned)(uintptr_t)(&sA[0][0]);
  const unsigned sW0 = (unsigned)(uintptr_t)(&sW[0][0]);
  unsigned long long gaA[4]; unsigned laA[4];
  unsigned long long gaW[2]; unsigned laW[2];
#pragma unroll
  for (int i = 0; i < 4; ++i) {
    const int idx = tid + i * 256;
    const int row = idx >> 3, c = idx & 7;
    gaA[i] = (unsigned long long)(uintptr_t)A + ((size_t)(m0 + row) * lda) * 2 + c * 16;
    laA[i] = sA0 + row * (LDTA * 2) + c * 16;
  }
#pragma unroll
  for (int i = 0; i < 2; ++i) {
    const int idx = tid + i * 256;
    const int row = idx >> 3, c = idx & 7;
    int gr = n0 + row; if (gr >= N) gr = N - 1;   // clamp partial-N (never stored)
    gaW[i] = (unsigned long long)(uintptr_t)W + ((size_t)gr * K) * 2 + c * 16;
    laW[i] = sW0 + row * (LDTA * 2) + c * 16;
  }

  v8f acc[2][2] = {};

  // prime buffer 0
#pragma unroll
  for (int i = 0; i < 4; ++i) { async_cp16(laA[i], gaA[i]); gaA[i] += BK * 2; }
#pragma unroll
  for (int i = 0; i < 2; ++i) { async_cp16(laW[i], gaW[i]); gaW[i] += BK * 2; }

  int cur = 0;
  for (int k0 = 0; k0 < K; k0 += BK) {
    if (k0 + BK < K) {
      const unsigned abuf = (cur ^ 1) ? A_TILE_B : 0u;
      const unsigned wbuf = (cur ^ 1) ? W_TILE_B : 0u;
#pragma unroll
      for (int i = 0; i < 4; ++i) { async_cp16(laA[i] + abuf, gaA[i]); gaA[i] += BK * 2; }
#pragma unroll
      for (int i = 0; i < 2; ++i) { async_cp16(laW[i] + wbuf, gaW[i]); gaW[i] += BK * 2; }
      // 6 newest (next tile) may stay in flight; oldest 6 (this tile) must land
      asm volatile("s_wait_asynccnt 0x6" ::: "memory");
    } else {
      asm volatile("s_wait_asynccnt 0x0" ::: "memory");
    }
    __syncthreads();

    const bf16* pA = &sA[cur][0];
    const bf16* pW = &sW[cur][0];
#pragma unroll
    for (int kc = 0; kc < 2; ++kc) {           // two 32-wide K chunks per tile
      v16bf afr[2], bfr[2];
#pragma unroll
      for (int i = 0; i < 2; ++i)
        afr[i] = load_fragA(pA + (wm * 32 + i * 16 + lrow) * LDTA + kc * 32, lhalf);
#pragma unroll
      for (int j = 0; j < 2; ++j)
        bfr[j] = load_fragB(pW + (wn * 32 + j * 16 + lrow) * LDTA + kc * 32, lhalf);
#pragma unroll
      for (int i = 0; i < 2; ++i)
#pragma unroll
        for (int j = 0; j < 2; ++j)
          acc[i][j] = __builtin_amdgcn_wmma_f32_16x16x32_bf16(
              false, afr[i], false, bfr[j], (short)0, acc[i][j], false, false);
    }
    __syncthreads();
    cur ^= 1;
  }

  // epilogue: C/D layout -> lane L holds col (L&15), VGPR r holds row 8*(L>>4)+r
#pragma unroll
  for (int j = 0; j < 2; ++j) {
    const int col = n0 + wn * 32 + j * 16 + lrow;
    const bool ok = (col < N);
    const float bv = ok ? bias[col] : 0.f;
#pragma unroll
    for (int i = 0; i < 2; ++i)
#pragma unroll
      for (int r = 0; r < 8; ++r) {
        const int row = m0 + wm * 32 + i * 16 + lhalf * 8 + r;
        float v = acc[i][j][r] + bv;
        if (RELU) v = fmaxf(v, 0.f);
        if (ok) C[(size_t)row * ldc + col] = (TOut)v;
      }
  }
}

// --------------------------- distributional tail ---------------------------
__global__ void zero_out_kernel(float* out) { if (threadIdx.x == 0) out[0] = 0.f; }

// q_atoms -> softmax(clip) -> <p,z> -> argmax over actions
__global__ __launch_bounds__(64)
void next_action_kernel(const float* __restrict__ val, const float* __restrict__ adv,
                        const float* __restrict__ z, int* __restrict__ nact) {
  const int row = blockIdx.x, tid = threadIdx.x;
  __shared__ float s_adv[OA];
  __shared__ float s_val[ATOMS], s_mean[ATOMS], s_z[ATOMS];
  __shared__ float s_q[NACT];

  if (tid < ATOMS) s_z[tid] = z[tid];
  for (int i = tid; i < OA; i += 64) s_adv[i] = adv[(size_t)row * OA + i];
  if (tid < ATOMS) s_val[tid] = val[(size_t)row * VAL_LD + tid];
  __syncthreads();
  if (tid < ATOMS) {
    float m = 0.f;
#pragma unroll
    for (int o = 0; o < NACT; ++o) m += s_adv[o * ATOMS + tid];
    s_mean[tid] = m * (1.0f / NACT);
  }
  __syncthreads();
  if (tid < NACT) {
    const int o = tid;
    float mx = -3.4e38f;
    for (int j = 0; j < ATOMS; ++j)
      mx = fmaxf(mx, s_val[j] + s_adv[o * ATOMS + j] - s_mean[j]);
    float sum = 0.f;
    for (int j = 0; j < ATOMS; ++j)
      sum += __expf(s_val[j] + s_adv[o * ATOMS + j] - s_mean[j] - mx);
    const float inv = 1.f / sum;
    float q = 0.f;
    for (int j = 0; j < ATOMS; ++j) {
      const float p = fmaxf(__expf(s_val[j] + s_adv[o * ATOMS + j] - s_mean[j] - mx) * inv, 0.001f);
      q += p * s_z[j];
    }
    s_q[o] = q;
  }
  __syncthreads();
  if (tid == 0) {
    int best = 0; float bv = s_q[0];
    for (int o = 1; o < NACT; ++o) if (s_q[o] > bv) { bv = s_q[o]; best = o; }
    nact[row] = best;
  }
}

// target dist at next_action, then categorical projection (delta_z = 4)
__global__ __launch_bounds__(64)
void project_kernel(const float* __restrict__ val, const float* __restrict__ adv,
                    const float* __restrict__ z, const float* __restrict__ r,
                    const int* __restrict__ nact, float* __restrict__ proj) {
  const int row = blockIdx.x, tid = threadIdx.x;
  __shared__ float s_adv[OA];
  __shared__ float s_val[ATOMS], s_mean[ATOMS], s_z[ATOMS];
  __shared__ float s_buf[ATOMS], s_nd[ATOMS], s_proj[ATOMS];
  __shared__ float s_red[2];

  const int na = nact[row];
  if (tid < ATOMS) s_z[tid] = z[tid];
  for (int i = tid; i < OA; i += 64) s_adv[i] = adv[(size_t)row * OA + i];
  if (tid < ATOMS) s_val[tid] = val[(size_t)row * VAL_LD + tid];
  __syncthreads();
  if (tid < ATOMS) {
    float m = 0.f;
#pragma unroll
    for (int o = 0; o < NACT; ++o) m += s_adv[o * ATOMS + tid];
    s_mean[tid] = m * (1.0f / NACT);
  }
  __syncthreads();
  if (tid < ATOMS) s_buf[tid] = s_val[tid] + s_adv[na * ATOMS + tid] - s_mean[tid];
  __syncthreads();
  if (tid == 0) {
    float mx = -3.4e38f;
    for (int j = 0; j < ATOMS; ++j) mx = fmaxf(mx, s_buf[j]);
    float sum = 0.f;
    for (int j = 0; j < ATOMS; ++j) sum += __expf(s_buf[j] - mx);
    s_red[0] = mx; s_red[1] = sum;
  }
  __syncthreads();
  if (tid < ATOMS) s_nd[tid] = fmaxf(__expf(s_buf[tid] - s_red[0]) / s_red[1], 0.001f);
  if (tid < ATOMS) s_proj[tid] = 0.f;
  __syncthreads();
  if (tid < ATOMS) {
    const float tz = fminf(fmaxf(r[row] + 0.9f * s_z[tid], 0.f), 200.f);
    const float b  = tz * 0.25f;
    const float fl = floorf(b), fu = ceilf(b);
    const float nd = s_nd[tid];
    atomicAdd(&s_proj[(int)fl], nd * (fu - b));
    atomicAdd(&s_proj[(int)fu], nd * (b - fl));
  }
  __syncthreads();
  if (tid < ATOMS) proj[(size_t)row * ATOMS + tid] = s_proj[tid];
}

// online log-prob at a[row]; loss accumulation
__global__ __launch_bounds__(64)
void loss_kernel(const float* __restrict__ val, const float* __restrict__ adv,
                 const float* __restrict__ proj, const int* __restrict__ act,
                 float* __restrict__ out) {
  const int row = blockIdx.x, tid = threadIdx.x;
  __shared__ float s_adv[OA];
  __shared__ float s_val[ATOMS], s_mean[ATOMS], s_buf[ATOMS], s_proj[ATOMS];

  const int a = act[row];
  for (int i = tid; i < OA; i += 64) s_adv[i] = adv[(size_t)row * OA + i];
  if (tid < ATOMS) s_val[tid] = val[(size_t)row * VAL_LD + tid];
  if (tid < ATOMS) s_proj[tid] = proj[(size_t)row * ATOMS + tid];
  __syncthreads();
  if (tid < ATOMS) {
    float m = 0.f;
#pragma unroll
    for (int o = 0; o < NACT; ++o) m += s_adv[o * ATOMS + tid];
    s_mean[tid] = m * (1.0f / NACT);
  }
  __syncthreads();
  if (tid < ATOMS) s_buf[tid] = s_val[tid] + s_adv[a * ATOMS + tid] - s_mean[tid];
  __syncthreads();
  if (tid == 0) {
    float mx = -3.4e38f;
    for (int j = 0; j < ATOMS; ++j) mx = fmaxf(mx, s_buf[j]);
    float sum = 0.f;
    for (int j = 0; j < ATOMS; ++j) sum += __expf(s_buf[j] - mx);
    float acc = 0.f;
    for (int j = 0; j < ATOMS; ++j) {
      const float p = fmaxf(__expf(s_buf[j] - mx) / sum, 0.001f);
      acc += s_proj[j] * __logf(p);
    }
    atomicAdd(out, -acc * (1.0f / BSZ));
  }
}

// ----------------------------- launcher ------------------------------------
extern "C" void kernel_launch(void* const* d_in, const int* in_sizes, int n_in,
                              void* d_out, int out_size, void* d_ws, size_t ws_size,
                              hipStream_t stream) {
  (void)in_sizes; (void)n_in; (void)out_size; (void)ws_size;

  const float* s_in = (const float*)d_in[0];
  const float* sp   = (const float*)d_in[1];
  const float* r    = (const float*)d_in[2];
  const float* z    = (const float*)d_in[3];
  const float* P[20];
  for (int i = 0; i < 20; ++i) P[i] = (const float*)d_in[4 + i];
  const int* act = (const int*)d_in[24];

  // ---- workspace carve-out (~94 MB)
  char* wp = (char*)d_ws;
  auto alloc = [&](size_t bytes) -> void* {
    void* q = (void*)wp;
    wp += (bytes + 255) & ~(size_t)255;
    return q;
  };
  bf16* sb  = (bf16*)alloc((size_t)BSZ * DIN * sizeof(bf16));
  bf16* spb = (bf16*)alloc((size_t)BSZ * DIN * sizeof(bf16));
  // per model: fw, vhw, ahw, vw, aw (biases stay f32, read directly)
  const size_t WSZ[5] = {(size_t)HID * DIN, (size_t)HID * HID, (size_t)HID * HID,
                         (size_t)ATOMS * HID, (size_t)OA * HID};
  bf16* wb[2][5];
  for (int m = 0; m < 2; ++m)
    for (int k = 0; k < 5; ++k)
      wb[m][k] = (bf16*)alloc(WSZ[k] * sizeof(bf16));
  bf16* feat = (bf16*)alloc((size_t)BSZ * HID * sizeof(bf16));
  bf16* vh   = (bf16*)alloc((size_t)BSZ * HID * sizeof(bf16));
  bf16* ah   = (bf16*)alloc((size_t)BSZ * HID * sizeof(bf16));
  float* valb = (float*)alloc((size_t)BSZ * VAL_LD * sizeof(float));
  float* advb = (float*)alloc((size_t)BSZ * OA * sizeof(float));
  int*   nact = (int*)alloc((size_t)BSZ * sizeof(int));
  float* proj = (float*)alloc((size_t)BSZ * ATOMS * sizeof(float));

  auto cvt = [&](const float* src, bf16* dst, size_t n) {
    const int n4 = (int)(n / 4);
    cvt_f32_to_bf16<<<dim3((n4 + 255) / 256), dim3(256), 0, stream>>>(src, dst, n4);
  };

  // ---- one-time conversions
  cvt(s_in, sb, (size_t)BSZ * DIN);
  cvt(sp, spb, (size_t)BSZ * DIN);
  const int WIDX[5] = {0, 2, 6, 4, 8};   // fw, vhw, ahw, vw, aw within a param set
  for (int m = 0; m < 2; ++m)
    for (int k = 0; k < 5; ++k)
      cvt(P[m * 10 + WIDX[k]], wb[m][k], WSZ[k]);

  // ---- three passes; pass 0: model(s'); pass 1: target(s'); pass 2: model(s)
  const bf16* X[3]  = {spb, spb, sb};
  const int   PM[3] = {0, 1, 0};         // weight set
  const int   PB[3] = {0, 10, 0};        // bias base within P[]

  const dim3 blk(256);
  const dim3 gHID(HID / BN, BSZ / BM);
  const dim3 gVAL(1, BSZ / BM);
  const dim3 gADV((OA + BN - 1) / BN, BSZ / BM);

  zero_out_kernel<<<1, 1, 0, stream>>>((float*)d_out);

  for (int i = 0; i < 3; ++i) {
    const int m = PM[i], pb = PB[i];
    gemm_bf16_async<bf16, true><<<gHID, blk, 0, stream>>>(
        X[i], DIN, wb[m][0], P[pb + 1], feat, HID, HID, DIN);
    gemm_bf16_async<bf16, true><<<gHID, blk, 0, stream>>>(
        feat, HID, wb[m][1], P[pb + 3], vh, HID, HID, HID);
    gemm_bf16_async<bf16, true><<<gHID, blk, 0, stream>>>(
        feat, HID, wb[m][2], P[pb + 7], ah, HID, HID, HID);
    gemm_bf16_async<float, false><<<gVAL, blk, 0, stream>>>(
        vh, HID, wb[m][3], P[pb + 5], valb, VAL_LD, ATOMS, HID);
    gemm_bf16_async<float, false><<<gADV, blk, 0, stream>>>(
        ah, HID, wb[m][4], P[pb + 9], advb, OA, OA, HID);

    if (i == 0)      next_action_kernel<<<BSZ, 64, 0, stream>>>(valb, advb, z, nact);
    else if (i == 1) project_kernel<<<BSZ, 64, 0, stream>>>(valb, advb, z, r, nact, proj);
    else             loss_kernel<<<BSZ, 64, 0, stream>>>(valb, advb, proj, act, (float*)d_out);
  }
}
